// MambaLayer_21492016349886
// MI455X (gfx1250) — matrix operand order repaired and checked by reading, not compiled, not run on hardware
//
#include <hip/hip_runtime.h>
#include <math.h>

// ---------------- problem constants ----------------
constexpr int LSEQ = 18432;           // 8*48*48
constexpr int TT = 8, HH = 48, WWD = 48;
constexpr int DIN = 128, DST = 16, DTR = 4, HID = 256;
constexpr int NCH = 96, CLEN = 192;   // scan chunking: 96*192 = 18432
constexpr int MT = LSEQ / 16;         // 1152 M-tiles
constexpr int DBS = 48;               // padded dbc row stride (36 -> 48)

typedef __attribute__((ext_vector_type(16))) _Float16 v16h;
typedef __attribute__((ext_vector_type(8)))  float    v8f;

__device__ __forceinline__ float siluf(float x) { return x / (1.f + expf(-x)); }
__device__ __forceinline__ float geluf(float x) { return 0.5f * x * (1.f + erff(x * 0.70710678118654752f)); }

__device__ __forceinline__ float4 ld4(const float* __restrict__ p) {
  return *reinterpret_cast<const float4*>(p);
}
__device__ __forceinline__ void cvt4(v16h& d, int o, float4 f) {
  d[o] = (_Float16)f.x; d[o + 1] = (_Float16)f.y;
  d[o + 2] = (_Float16)f.z; d[o + 3] = (_Float16)f.w;
}

// ---------------- WMMA GEMM, compile-time K: C[M x N] = A[M x K] @ W[N x K]^T (+bias) ----------------
// A layout per lane: two contiguous 8-float segments [kb+8h, +8) and [kb+16+8h, +8).
// B layout per lane: two contiguous 8-float segments [kb+8h, +8) and [kb+16+8h, +8).
template <int K>
__global__ __launch_bounds__(256) void gemm_wmma(const float* __restrict__ A, int lda,
                                                 const float* __restrict__ Wm, int ldw,
                                                 const float* __restrict__ bias,
                                                 float* __restrict__ C, int ldc,
                                                 int Mtiles, int tiles_n) {
  const int wave = threadIdx.x >> 5, lane = threadIdx.x & 31;
  const int tile = blockIdx.x * 8 + wave;
  if (tile >= Mtiles * tiles_n) return;
  const int tm = tile / tiles_n, tn = tile % tiles_n;
  const int ml = lane & 15, half = lane >> 4;
  const int n = tn * 16 + ml;
  const float* arow = A + (size_t)(tm * 16 + ml) * lda;
  const float* wrow = Wm + (size_t)n * ldw;
  v8f acc = {};
#pragma unroll
  for (int kb = 0; kb < K; kb += 32) {
    const float* ap = arow + kb + 8 * half;
    const float* bp = wrow + kb + 8 * half;
    const float4 a0 = ld4(ap), a1 = ld4(ap + 4), a2 = ld4(ap + 16), a3 = ld4(ap + 20);
    const float4 b0 = ld4(bp), b1 = ld4(bp + 4), b2 = ld4(bp + 16), b3 = ld4(bp + 20);
    v16h a, b;
    cvt4(a, 0, a0); cvt4(a, 4, a1); cvt4(a, 8, a2); cvt4(a, 12, a3);
    cvt4(b, 0, b0); cvt4(b, 4, b1); cvt4(b, 8, b2); cvt4(b, 12, b3);
    acc = __builtin_amdgcn_wmma_f32_16x16x32_f16(false, a, false, b, (short)0, acc, false, false);
  }
  const float bv = bias ? bias[n] : 0.f;
#pragma unroll
  for (int r = 0; r < 8; ++r)
    C[(size_t)(tm * 16 + r + 8 * half) * ldc + n] = acc[r] + bv;
}

// ---------------- implicit-GEMM 3x3x3 dense conv via WMMA (branchless gather) ----------------
template <int Cin>
__device__ __forceinline__ float im2col3(const float* __restrict__ src,
                                         int t0, int h0, int w0, int k) {
  const int c = k / 27, r = k % 27;
  const int dt = r / 9, dh = (r % 9) / 3, dw = r % 3;
  const int t = t0 + dt - 1, h = h0 + dh - 1, w = w0 + dw - 1;
  const bool inb = ((unsigned)t < (unsigned)TT) & ((unsigned)h < (unsigned)HH) &
                   ((unsigned)w < (unsigned)WWD) & (c < Cin);
  const size_t idx = inb ? ((size_t)((t * HH + h) * WWD + w) * Cin + c) : 0;
  const float v = src[idx];           // address 0 always valid; value masked below
  return inb ? v : 0.f;
}

// Wm is pre-padded to [Nout][KPAD] with zeros (KPAD multiple of 32).
template <int Cin, int KPAD>
__global__ __launch_bounds__(256) void conv3d_gemm_wmma(const float* __restrict__ src,
                                                        const float* __restrict__ Wm,
                                                        const float* __restrict__ bias,
                                                        float* __restrict__ dst, int Nout,
                                                        int tiles_n) {
  const int wave = threadIdx.x >> 5, lane = threadIdx.x & 31;
  const int tile = blockIdx.x * 8 + wave;
  if (tile >= MT * tiles_n) return;
  const int tm = tile / tiles_n, tn = tile % tiles_n;
  const int ml = lane & 15, half = lane >> 4;
  const int l = tm * 16 + ml;
  const int t0 = l / (HH * WWD), rem = l % (HH * WWD), h0 = rem / WWD, w0 = rem % WWD;
  const int n = tn * 16 + ml;
  const float* wrow = Wm + (size_t)n * KPAD;
  v8f acc = {};
  for (int kb = 0; kb < KPAD; kb += 32) {
    const int ka0 = kb + 8 * half;        // A segment 0: ka0..ka0+7
    const int ka1 = kb + 16 + 8 * half;   // A segment 1
    const float* bp = wrow + kb + 8 * half;
    const float4 b0 = ld4(bp), b1 = ld4(bp + 4), b2 = ld4(bp + 16), b3 = ld4(bp + 20);
    v16h a, b;
#pragma unroll
    for (int j = 0; j < 8; ++j) {
      a[j]     = (_Float16)im2col3<Cin>(src, t0, h0, w0, ka0 + j);
      a[8 + j] = (_Float16)im2col3<Cin>(src, t0, h0, w0, ka1 + j);
    }
    cvt4(b, 0, b0); cvt4(b, 4, b1); cvt4(b, 8, b2); cvt4(b, 12, b3);
    acc = __builtin_amdgcn_wmma_f32_16x16x32_f16(false, a, false, b, (short)0, acc, false, false);
  }
  const float bv = bias[n];
#pragma unroll
  for (int r = 0; r < 8; ++r)
    dst[(size_t)(tm * 16 + r + 8 * half) * Nout + n] = acc[r] + bv;
}

// ---------------- weight padding: dst[r][k] (drows x dcols) from src (srows x scols) ----------------
__global__ void pad2d_kernel(const float* __restrict__ src, float* __restrict__ dst,
                             int srows, int scols, int drows, int dcols) {
  const int i = blockIdx.x * blockDim.x + threadIdx.x;
  if (i >= drows * dcols) return;
  const int r = i / dcols, k = i % dcols;
  dst[i] = (r < srows && k < scols) ? src[r * scols + k] : 0.f;
}

// ---------------- LayerNorm over 64 channels (strided src, dense L x 64 dst) ----------------
__global__ void ln64_kernel(const float* __restrict__ src, long sl, long sc,
                            const float* __restrict__ g, const float* __restrict__ b,
                            float* __restrict__ dst) {
  const int l = blockIdx.x * blockDim.x + threadIdx.x;
  if (l >= LSEQ) return;
  const float* p = src + (size_t)l * sl;
  float s = 0.f, ss = 0.f;
  for (int c = 0; c < 64; ++c) { const float v = p[(size_t)c * sc]; s += v; ss += v * v; }
  const float m = s * (1.f / 64.f);
  const float var = ss * (1.f / 64.f) - m * m;
  const float inv = rsqrtf(var + 1e-5f);
  for (int c = 0; c < 64; ++c) {
    const float v = p[(size_t)c * sc];
    dst[(size_t)l * 64 + c] = (v - m) * inv * g[c] + b[c];
  }
}

// ---------------- causal depthwise conv1d (k=4) + silu, optional sequence reversal ----------------
__global__ void conv1d_silu_kernel(const float* __restrict__ xz, const float* __restrict__ w,
                                   const float* __restrict__ b, float* __restrict__ xc, int rev) {
  const int idx = blockIdx.x * blockDim.x + threadIdx.x;
  if (idx >= LSEQ * DIN) return;
  const int p = idx >> 7, d = idx & 127;
  float acc = b[d];
#pragma unroll
  for (int j = 0; j < 4; ++j) {
    const int q = p - 3 + j;
    if (q >= 0) {
      const int l = rev ? (LSEQ - 1 - q) : q;
      acc += w[d * 4 + j] * xz[(size_t)l * 256 + d];
    }
  }
  xc[idx] = siluf(acc);
}

// ---------------- dt = softplus(dbc[:, :4] @ dt_proj^T + bias)  (dbc stride = DBS) ----------------
__global__ void dt_kernel(const float* __restrict__ dbc, const float* __restrict__ dtw,
                          const float* __restrict__ dtb, float* __restrict__ dt) {
  const int idx = blockIdx.x * blockDim.x + threadIdx.x;
  if (idx >= LSEQ * DIN) return;
  const int p = idx >> 7, d = idx & 127;
  float x = dtb[d];
#pragma unroll
  for (int r = 0; r < DTR; ++r) x += dbc[(size_t)p * DBS + r] * dtw[d * DTR + r];
  dt[idx] = (x > 20.f) ? x : log1pf(expf(x));
}

// ---------------- selective scan: 3-pass chunked scan ----------------
__global__ void scan_pass1(const float* __restrict__ dt, const float* __restrict__ xc,
                           const float* __restrict__ dbc, const float* __restrict__ A_log,
                           float* __restrict__ prodA, float* __restrict__ hend) {
  const int idx = blockIdx.x * blockDim.x + threadIdx.x;
  if (idx >= DIN * NCH) return;
  const int d = idx / NCH, ch = idx % NCH;
  float A[DST], h[DST], pr[DST];
#pragma unroll
  for (int n = 0; n < DST; ++n) { A[n] = -expf(A_log[d * DST + n]); h[n] = 0.f; pr[n] = 1.f; }
  const int p0 = ch * CLEN;
  for (int i = 0; i < CLEN; ++i) {
    const int p = p0 + i;
    const float dtv = dt[(size_t)p * DIN + d];
    const float u = xc[(size_t)p * DIN + d];
    const float du = dtv * u;
    const float* brow = dbc + (size_t)p * DBS + DTR;
#pragma unroll
    for (int n = 0; n < DST; ++n) {
      const float dA = expf(dtv * A[n]);
      h[n] = dA * h[n] + du * brow[n];
      pr[n] *= dA;
    }
  }
  const size_t base = ((size_t)d * NCH + ch) * DST;
#pragma unroll
  for (int n = 0; n < DST; ++n) { prodA[base + n] = pr[n]; hend[base + n] = h[n]; }
}

__global__ void scan_pass2(const float* __restrict__ prodA, const float* __restrict__ hend,
                           float* __restrict__ carry) {
  const int tid = threadIdx.x;            // tid = branch*128 + d, 256 threads
  float c[DST];
#pragma unroll
  for (int n = 0; n < DST; ++n) c[n] = 0.f;
  const size_t chain = (size_t)tid * NCH * DST;
  for (int j = 0; j < NCH; ++j) {
    const size_t base = chain + (size_t)j * DST;
#pragma unroll
    for (int n = 0; n < DST; ++n) {
      carry[base + n] = c[n];
      c[n] = prodA[base + n] * c[n] + hend[base + n];
    }
  }
}

__global__ void scan_pass3(const float* __restrict__ dt, const float* __restrict__ xc,
                           const float* __restrict__ dbc, const float* __restrict__ A_log,
                           const float* __restrict__ Dp, const float* __restrict__ carry,
                           const float* __restrict__ xz, float* __restrict__ ytot, int rev) {
  const int idx = blockIdx.x * blockDim.x + threadIdx.x;
  if (idx >= DIN * NCH) return;
  const int d = idx / NCH, ch = idx % NCH;
  float A[DST], h[DST];
  const size_t base = ((size_t)d * NCH + ch) * DST;
#pragma unroll
  for (int n = 0; n < DST; ++n) { A[n] = -expf(A_log[d * DST + n]); h[n] = carry[base + n]; }
  const float Dv = Dp[d];
  const int p0 = ch * CLEN;
  for (int i = 0; i < CLEN; ++i) {
    const int p = p0 + i;
    const float dtv = dt[(size_t)p * DIN + d];
    const float u = xc[(size_t)p * DIN + d];
    const float du = dtv * u;
    const float* brow = dbc + (size_t)p * DBS + DTR;
    const float* crow = dbc + (size_t)p * DBS + DTR + DST;
    float y = 0.f;
#pragma unroll
    for (int n = 0; n < DST; ++n) {
      const float dA = expf(dtv * A[n]);
      h[n] = dA * h[n] + du * brow[n];
      y += h[n] * crow[n];
    }
    y += u * Dv;
    const int l = rev ? (LSEQ - 1 - p) : p;
    const float z = xz[(size_t)l * 256 + DIN + d];
    atomicAdd(&ytot[(size_t)l * DIN + d], y * siluf(z));
  }
}

// ---------------- elementwise helpers ----------------
__global__ void zero_kernel(float* __restrict__ p, int n) {
  const int i = blockIdx.x * blockDim.x + threadIdx.x;
  if (i < n) p[i] = 0.f;
}

__global__ void xm_kernel(const float* __restrict__ x, const float* __restrict__ opj,
                          const float* __restrict__ skip, float* __restrict__ xm) {
  const int idx = blockIdx.x * blockDim.x + threadIdx.x;
  if (idx >= LSEQ * 64) return;
  const int l = idx >> 6, c = idx & 63;
  xm[idx] = x[(size_t)c * LSEQ + l] * skip[0] + opj[idx];
}

__global__ void scale_add_kernel(const float* __restrict__ a, const float* __restrict__ s,
                                 const float* __restrict__ b, float* __restrict__ o, int n) {
  const int i = blockIdx.x * blockDim.x + threadIdx.x;
  if (i < n) o[i] = a[i] * s[0] + b[i];
}

__global__ void gelu_kernel(const float* __restrict__ a, float* __restrict__ o, int n) {
  const int i = blockIdx.x * blockDim.x + threadIdx.x;
  if (i < n) o[i] = geluf(a[i]);
}

// depthwise 3x3x3 conv over (HID=256, T, H, W), channel-last storage [L][256]
__global__ void dwconv3d_kernel(const float* __restrict__ src, const float* __restrict__ w,
                                const float* __restrict__ b, float* __restrict__ dst) {
  const int idx = blockIdx.x * blockDim.x + threadIdx.x;
  if (idx >= LSEQ * HID) return;
  const int l = idx >> 8, c = idx & 255;
  const int t0 = l / (HH * WWD), rem = l % (HH * WWD), h0 = rem / WWD, w0 = rem % WWD;
  float acc = b[c];
#pragma unroll
  for (int kd = 0; kd < 3; ++kd) {
    const int t = t0 + kd - 1;
    if ((unsigned)t >= (unsigned)TT) continue;
#pragma unroll
    for (int kh = 0; kh < 3; ++kh) {
      const int h = h0 + kh - 1;
      if ((unsigned)h >= (unsigned)HH) continue;
#pragma unroll
      for (int kw = 0; kw < 3; ++kw) {
        const int ww = w0 + kw - 1;
        if ((unsigned)ww >= (unsigned)WWD) continue;
        const int l2 = (t * HH + h) * WWD + ww;
        acc += w[c * 27 + kd * 9 + kh * 3 + kw] * src[(size_t)l2 * HID + c];
      }
    }
  }
  dst[idx] = acc;
}

__global__ void pool_kernel(const float* __restrict__ cb2, float* __restrict__ pool) {
  __shared__ float sd[256];
  const int c = blockIdx.x;
  float s = 0.f;
  for (int l = threadIdx.x; l < LSEQ; l += 256) s += cb2[(size_t)l * 64 + c];
  sd[threadIdx.x] = s;
  __syncthreads();
  for (int st = 128; st > 0; st >>= 1) {
    if ((int)threadIdx.x < st) sd[threadIdx.x] += sd[threadIdx.x + st];
    __syncthreads();
  }
  if (threadIdx.x == 0) pool[c] = sd[0] / (float)LSEQ;
}

__global__ void att_kernel(const float* __restrict__ pool,
                           const float* __restrict__ w1, const float* __restrict__ b1,
                           const float* __restrict__ w2, const float* __restrict__ b2,
                           float* __restrict__ att) {
  __shared__ float s1[8];
  const int t = threadIdx.x;
  if (t < 8) {
    float s = b1[t];
    for (int c = 0; c < 64; ++c) s += w1[t * 64 + c] * pool[c];
    s1[t] = fmaxf(s, 0.f);
  }
  __syncthreads();
  if (t < 64) {
    float s = b2[t];
#pragma unroll
    for (int j = 0; j < 8; ++j) s += w2[t * 8 + j] * s1[j];
    att[t] = 1.f / (1.f + expf(-s));
  }
}

__global__ void final_kernel(const float* __restrict__ obuf, const float* __restrict__ cb2,
                             const float* __restrict__ att, const float* __restrict__ skip,
                             float* __restrict__ out) {
  const int idx = blockIdx.x * blockDim.x + threadIdx.x;
  if (idx >= LSEQ * 64) return;
  const int l = idx >> 6, c = idx & 63;
  out[(size_t)c * LSEQ + l] = obuf[idx] * skip[0] + cb2[idx] * att[c];
}

// ---------------- host orchestration ----------------
static inline int cdiv(int a, int b) { return (a + b - 1) / b; }
static inline int gemm_blocks(int Mtiles, int tiles_n) { return cdiv(Mtiles * tiles_n, 8); }

extern "C" void kernel_launch(void* const* d_in, const int* in_sizes, int n_in,
                              void* d_out, int out_size, void* d_ws, size_t ws_size,
                              hipStream_t stream) {
  (void)in_sizes; (void)n_in; (void)out_size; (void)ws_size;
  const float* x         = (const float*)d_in[0];
  const float* in_proj_w = (const float*)d_in[1];
  const float* c1w_f     = (const float*)d_in[2];
  const float* c1b_f     = (const float*)d_in[3];
  const float* xpw_f     = (const float*)d_in[4];
  const float* dtw_f     = (const float*)d_in[5];
  const float* dtb_f     = (const float*)d_in[6];
  const float* Alog_f    = (const float*)d_in[7];
  const float* Dp_f      = (const float*)d_in[8];
  const float* c1w_b     = (const float*)d_in[9];
  const float* c1b_b     = (const float*)d_in[10];
  const float* xpw_b     = (const float*)d_in[11];
  const float* dtw_b     = (const float*)d_in[12];
  const float* dtb_b     = (const float*)d_in[13];
  const float* Alog_b    = (const float*)d_in[14];
  const float* Dp_b      = (const float*)d_in[15];
  const float* out_proj  = (const float*)d_in[16];
  const float* ln1w      = (const float*)d_in[17];
  const float* ln1b      = (const float*)d_in[18];
  const float* ln2w      = (const float*)d_in[19];
  const float* ln2b      = (const float*)d_in[20];
  const float* ln3w      = (const float*)d_in[21];
  const float* ln3b      = (const float*)d_in[22];
  const float* skip1     = (const float*)d_in[23];
  const float* skip2     = (const float*)d_in[24];
  const float* skip3     = (const float*)d_in[25];
  const float* proj_w    = (const float*)d_in[26];
  const float* proj_b    = (const float*)d_in[27];
  const float* fc1w      = (const float*)d_in[28];
  const float* fc1b      = (const float*)d_in[29];
  const float* dww       = (const float*)d_in[30];
  const float* dwb       = (const float*)d_in[31];
  const float* fc2w      = (const float*)d_in[32];
  const float* fc2b      = (const float*)d_in[33];
  const float* cab1w     = (const float*)d_in[34];
  const float* cab1b     = (const float*)d_in[35];
  const float* cab2w     = (const float*)d_in[36];
  const float* cab2b     = (const float*)d_in[37];
  const float* ca1w      = (const float*)d_in[38];
  const float* ca1b      = (const float*)d_in[39];
  const float* ca2w      = (const float*)d_in[40];
  const float* ca2b      = (const float*)d_in[41];
  float* out = (float*)d_out;

  // workspace layout (floats)
  float* ws = (float*)d_ws;
  size_t off = 0;
  float* xn    = ws + off; off += (size_t)LSEQ * 64;
  float* xz    = ws + off; off += (size_t)LSEQ * 256;
  float* xc_f  = ws + off; off += (size_t)LSEQ * DIN;
  float* xc_b  = ws + off; off += (size_t)LSEQ * DIN;
  float* dbc_f = ws + off; off += (size_t)LSEQ * DBS;
  float* dbc_b = ws + off; off += (size_t)LSEQ * DBS;
  float* dt_f  = ws + off; off += (size_t)LSEQ * DIN;
  float* dt_b  = ws + off; off += (size_t)LSEQ * DIN;
  const size_t plane = (size_t)DIN * NCH * DST;        // per-branch scan plane
  float* prodA = ws + off; off += 2 * plane;
  float* hend  = ws + off; off += 2 * plane;
  float* carry = ws + off; off += 2 * plane;
  float* ytot  = ws + off; off += (size_t)LSEQ * DIN;
  float* opj   = ws + off; off += (size_t)LSEQ * 64;
  float* xm    = ws + off; off += (size_t)LSEQ * 64;
  float* t64   = ws + off; off += (size_t)LSEQ * 64;   // ln2 out, later ln3 out
  float* h1    = ws + off; off += (size_t)LSEQ * HID;  // fc1 out, later gelu(h2)
  float* h2    = ws + off; off += (size_t)LSEQ * HID;
  float* ffn   = ws + off; off += (size_t)LSEQ * 64;
  float* xm2   = ws + off; off += (size_t)LSEQ * 64;
  float* obuf  = ws + off; off += (size_t)LSEQ * 64;
  float* cb1   = ws + off; off += (size_t)LSEQ * 16;
  float* cb1g  = ws + off; off += (size_t)LSEQ * 16;
  float* cb2   = ws + off; off += (size_t)LSEQ * 64;
  float* pool  = ws + off; off += 64;
  float* att   = ws + off; off += 64;
  float* xpwp_f = ws + off; off += (size_t)DBS * DIN;  // x_proj_w padded 48x128
  float* xpwp_b = ws + off; off += (size_t)DBS * DIN;
  float* cab2wp = ws + off; off += (size_t)64 * 448;   // cab2_w padded 64x448

  const int TPB = 256;
  const int gL    = cdiv(LSEQ, TPB);
  const int gLD   = cdiv(LSEQ * DIN, TPB);
  const int gL64  = cdiv(LSEQ * 64, TPB);
  const int gLH   = cdiv(LSEQ * HID, TPB);
  const int gScan = cdiv(DIN * NCH, TPB);

  // 0) zero-padded weight copies (x_proj 36x128 -> 48x128; cab2 64x432 -> 64x448)
  pad2d_kernel<<<cdiv(DBS * DIN, TPB), TPB, 0, stream>>>(xpw_f, xpwp_f, 36, DIN, DBS, DIN);
  pad2d_kernel<<<cdiv(DBS * DIN, TPB), TPB, 0, stream>>>(xpw_b, xpwp_b, 36, DIN, DBS, DIN);
  pad2d_kernel<<<cdiv(64 * 448, TPB), TPB, 0, stream>>>(cab2w, cab2wp, 64, 432, 64, 448);

  // 1) LN1 (input channel-first [64][L]) -> xn [L][64]
  ln64_kernel<<<gL, TPB, 0, stream>>>(x, 1L, (long)LSEQ, ln1w, ln1b, xn);
  // 2) in_proj: xz = xn @ in_proj_w^T   (L x 256)
  gemm_wmma<64><<<gemm_blocks(MT, 16), TPB, 0, stream>>>(xn, 64, in_proj_w, 64, nullptr, xz, 256, MT, 16);
  // 3) per-branch causal dwconv1d + silu (reverse branch scans backwards)
  conv1d_silu_kernel<<<gLD, TPB, 0, stream>>>(xz, c1w_f, c1b_f, xc_f, 0);
  conv1d_silu_kernel<<<gLD, TPB, 0, stream>>>(xz, c1w_b, c1b_b, xc_b, 1);
  // 4) x_proj: dbc = xc @ x_proj_w^T  (L x 48, cols 36..47 are zero-weight pad)
  gemm_wmma<128><<<gemm_blocks(MT, 3), TPB, 0, stream>>>(xc_f, DIN, xpwp_f, DIN, nullptr, dbc_f, DBS, MT, 3);
  gemm_wmma<128><<<gemm_blocks(MT, 3), TPB, 0, stream>>>(xc_b, DIN, xpwp_b, DIN, nullptr, dbc_b, DBS, MT, 3);
  // 5) dt = softplus(dbc[:, :4] @ dt_proj^T + b)
  dt_kernel<<<gLD, TPB, 0, stream>>>(dbc_f, dtw_f, dtb_f, dt_f);
  dt_kernel<<<gLD, TPB, 0, stream>>>(dbc_b, dtw_b, dtb_b, dt_b);
  // 6) chunked selective scan (3 passes), both branches accumulate into ytot
  zero_kernel<<<gLD, TPB, 0, stream>>>(ytot, LSEQ * DIN);
  scan_pass1<<<gScan, TPB, 0, stream>>>(dt_f, xc_f, dbc_f, Alog_f, prodA,         hend);
  scan_pass1<<<gScan, TPB, 0, stream>>>(dt_b, xc_b, dbc_b, Alog_b, prodA + plane, hend + plane);
  scan_pass2<<<1, 256, 0, stream>>>(prodA, hend, carry);
  scan_pass3<<<gScan, TPB, 0, stream>>>(dt_f, xc_f, dbc_f, Alog_f, Dp_f, carry,         xz, ytot, 0);
  scan_pass3<<<gScan, TPB, 0, stream>>>(dt_b, xc_b, dbc_b, Alog_b, Dp_b, carry + plane, xz, ytot, 1);
  // 7) out_proj + residual skip1
  gemm_wmma<128><<<gemm_blocks(MT, 4), TPB, 0, stream>>>(ytot, DIN, out_proj, DIN, nullptr, opj, 64, MT, 4);
  xm_kernel<<<gL64, TPB, 0, stream>>>(x, opj, skip1, xm);
  // 8) MLP with depthwise conv3d
  ln64_kernel<<<gL, TPB, 0, stream>>>(xm, 64L, 1L, ln2w, ln2b, t64);
  gemm_wmma<64><<<gemm_blocks(MT, 16), TPB, 0, stream>>>(t64, 64, fc1w, 64, fc1b, h1, HID, MT, 16);
  dwconv3d_kernel<<<gLH, TPB, 0, stream>>>(h1, dww, dwb, h2);
  gelu_kernel<<<gLH, TPB, 0, stream>>>(h2, h1, LSEQ * HID);
  gemm_wmma<256><<<gemm_blocks(MT, 4), TPB, 0, stream>>>(h1, HID, fc2w, HID, fc2b, ffn, 64, MT, 4);
  scale_add_kernel<<<gL64, TPB, 0, stream>>>(xm, skip2, ffn, xm2, LSEQ * 64);
  // 9) output projection
  gemm_wmma<64><<<gemm_blocks(MT, 4), TPB, 0, stream>>>(xm2, 64, proj_w, 64, proj_b, obuf, 64, MT, 4);
  // 10) CAB: LN3 -> conv(64->16) -> gelu -> conv(16->64) -> channel attention
  ln64_kernel<<<gL, TPB, 0, stream>>>(obuf, 64L, 1L, ln3w, ln3b, t64);
  conv3d_gemm_wmma<64, 1728><<<gemm_blocks(MT, 1), TPB, 0, stream>>>(t64, cab1w, cab1b, cb1, 16, 1);
  gelu_kernel<<<cdiv(LSEQ * 16, TPB), TPB, 0, stream>>>(cb1, cb1g, LSEQ * 16);
  conv3d_gemm_wmma<16, 448><<<gemm_blocks(MT, 4), TPB, 0, stream>>>(cb1g, cab2wp, cab2b, cb2, 64, 4);
  pool_kernel<<<64, 256, 0, stream>>>(cb2, pool);
  att_kernel<<<1, 64, 0, stream>>>(pool, ca1w, ca1b, ca2w, ca2b, att);
  // 11) final: out[c][l] = obuf*skip3 + cb2*att  (channel-first output)
  final_kernel<<<gL64, TPB, 0, stream>>>(obuf, cb2, att, skip3, out);
}
